// HyperGraphFixedCapacity_81466939670770
// MI455X (gfx1250) — matrix operand order repaired and checked by reading, not compiled, not run on hardware
//
#include <hip/hip_runtime.h>
#include <hip/hip_bf16.h>
#include <math.h>

typedef __attribute__((ext_vector_type(16))) _Float16 v16h;
typedef __attribute__((ext_vector_type(8)))  float    v8f;

#define B_   2
#define N_   4096
#define D_   1024
#define H_   16
#define HD_  64
#define K_   4
#define CAP_ 1024
#define NEG_  (-1.0e9f)
#define SCALE_ 0.125f   /* 64^-0.5 */

#define WMMA_F16(a, b, c) \
  __builtin_amdgcn_wmma_f32_16x16x32_f16(false, (a), false, (b), (short)0, (c), false, false)

// ---------------------------------------------------------------------------
// Router: logits = x @ Wr (K=4), softmax, argmax, aux-loss accumulators.
// One wave per token.
// ---------------------------------------------------------------------------
__global__ void router_kernel(const float* __restrict__ x, const float* __restrict__ Wr,
                              float* __restrict__ probs, int* __restrict__ choice,
                              float* __restrict__ cnt_bk, float* __restrict__ psum_bk) {
  int wave = threadIdx.x >> 5, lane = threadIdx.x & 31;
  int tok = blockIdx.x * (blockDim.x >> 5) + wave;
  if (tok >= B_ * N_) return;
  const float* xr = x + (size_t)tok * D_;
  float acc[K_] = {0.f, 0.f, 0.f, 0.f};
  for (int d = lane; d < D_; d += 32) {
    float xv = xr[d];
#pragma unroll
    for (int e = 0; e < K_; ++e) acc[e] += xv * Wr[d * K_ + e];
  }
#pragma unroll
  for (int off = 16; off; off >>= 1)
#pragma unroll
    for (int e = 0; e < K_; ++e) acc[e] += __shfl_xor(acc[e], off, 32);

  float mx = fmaxf(fmaxf(acc[0], acc[1]), fmaxf(acc[2], acc[3]));
  float ex[K_], s = 0.f;
#pragma unroll
  for (int e = 0; e < K_; ++e) { ex[e] = __expf(acc[e] - mx); s += ex[e]; }
  int ch = 0; float best = acc[0];
#pragma unroll
  for (int e = 1; e < K_; ++e) if (acc[e] > best) { best = acc[e]; ch = e; }

  if (lane == 0) {
    int b = tok / N_;
#pragma unroll
    for (int e = 0; e < K_; ++e) {
      float p = ex[e] / s;
      probs[(size_t)tok * K_ + e] = p;
      atomicAdd(&psum_bk[b * K_ + e], p);
    }
    choice[tok] = ch;
    atomicAdd(&cnt_bk[b * K_ + ch], 1.0f);
  }
}

// ---------------------------------------------------------------------------
// Top-k: one block per (b,k). Bitonic sort of 4096 keys in LDS (32 KB).
// key = (monotonic(score) << 32) | ~idx  -> descending score, ascending idx.
// ---------------------------------------------------------------------------
__device__ __forceinline__ unsigned ord_f32(float f) {
  unsigned u = __float_as_uint(f);
  return (u >> 31) ? ~u : (u | 0x80000000u);
}

__global__ void __launch_bounds__(1024) topk_sort_kernel(
    const float* __restrict__ probs, const int* __restrict__ choice,
    int* __restrict__ topk, int* __restrict__ Lbuf) {
  __shared__ unsigned long long keys[N_];
  int bk = blockIdx.x, b = bk / K_, k = bk % K_;
  int tid = threadIdx.x;

  for (int n = tid; n < N_; n += 1024) {
    int tok = b * N_ + n;
    float sc = (choice[tok] == k) ? probs[(size_t)tok * K_ + k] : NEG_;
    keys[n] = ((unsigned long long)ord_f32(sc) << 32) | (unsigned)(~n);
  }
  __syncthreads();

  for (int size = 2; size <= N_; size <<= 1) {
    for (int stride = size >> 1; stride > 0; stride >>= 1) {
#pragma unroll 1
      for (int p = 0; p < N_ / 2; p += 1024) {
        int t = tid + p;
        int pos = 2 * t - (t & (stride - 1));
        unsigned long long a = keys[pos], c = keys[pos + stride];
        bool desc = ((pos & size) == 0);
        if (desc ? (a < c) : (a > c)) { keys[pos] = c; keys[pos + stride] = a; }
      }
      __syncthreads();
    }
  }

  if (tid < CAP_) {
    unsigned long long key = keys[tid];
    topk[bk * CAP_ + tid] = (int)(~(unsigned)(key & 0xFFFFFFFFu));
    if ((unsigned)(key >> 32) > ord_f32(-1.0e8f))
      atomicMax(&Lbuf[bk], tid + 1);
  }
}

// ---------------------------------------------------------------------------
// Gathered QKV projection GEMM (f32 A via gather, f32 W staged in LDS,
// f16 out). KT=false: (b,k,h,c,hd) layout.  KT=true: (b,k,h,hd,c) layout
// (pre-transposed K so attention B-fragments are contiguous loads).
// Wave tile 16x64, k-step 32, 4 WMMAs/iter. grid=(16, M/128), block=256.
// ---------------------------------------------------------------------------
template <bool KT>
__global__ void __launch_bounds__(256) qkv_gather_gemm(
    const float* __restrict__ x, const float* __restrict__ W,
    const int* __restrict__ topk, _Float16* __restrict__ outg) {
  __shared__ float wtile[32][65];   // padded: fragment reads hit distinct banks
  int wave = threadIdx.x >> 5, lane = threadIdx.x & 31;
  int m = lane & 15, half = lane >> 4;
  int row0 = blockIdx.y * 128 + wave * 16;
  int col0 = blockIdx.x * 64;

  int R = row0 + m;
  int token = topk[R];
  int b = R / (K_ * CAP_);
  const float* arow = x + ((size_t)b * N_ + token) * D_;

  v8f acc[4] = {};
  for (int kk = 0; kk < D_; kk += 32) {
    {  // cooperative W tile: 256 threads x 8 floats = 32x64
      int r  = threadIdx.x >> 3;
      int c8 = (threadIdx.x & 7) * 8;
      const float* wp = W + (size_t)(kk + r) * (H_ * HD_) + col0 + c8;
      __builtin_prefetch(wp + 32 * (H_ * HD_), 0, 0);
#pragma unroll
      for (int i = 0; i < 8; ++i) wtile[r][c8 + i] = wp[i];
    }
    __syncthreads();

    v16h a;
    const float* ap = arow + kk + (half ? 8 : 0);
#pragma unroll
    for (int i = 0; i < 8; ++i) { a[i] = (_Float16)ap[i]; a[i + 8] = (_Float16)ap[i + 16]; }

#pragma unroll
    for (int n = 0; n < 4; ++n) {
      v16h bf;
#pragma unroll
      for (int i = 0; i < 16; ++i) bf[i] = (_Float16)wtile[half * 8 + (i & 7) + (i >> 3) * 16][n * 16 + m];
      acc[n] = WMMA_F16(a, bf, acc[n]);
    }
    __syncthreads();
  }

#pragma unroll
  for (int n = 0; n < 4; ++n) {
#pragma unroll
    for (int r = 0; r < 8; ++r) {
      int Rr  = row0 + r + half * 8;
      int bb  = Rr / (K_ * CAP_), rem = Rr % (K_ * CAP_);
      int ke  = rem / CAP_, c = rem % CAP_;
      int col = col0 + n * 16 + m;
      int h = col >> 6, hd = col & 63;
      size_t head = ((size_t)bb * K_ + ke) * H_ + h;
      if (KT)  // transposed: consecutive r -> consecutive c -> coalesced stores
        outg[(head * HD_ + hd) * CAP_ + c] = (_Float16)acc[n][r];
      else
        outg[(head * CAP_ + c) * HD_ + hd] = (_Float16)acc[n][r];
    }
  }
}

// ---------------------------------------------------------------------------
// RoPE in-place: q in (head,c,hd), k in (head,hd,c). One thread per
// (head,c,i) rotates the (i, i+32) pair for both q and k.
// ---------------------------------------------------------------------------
__global__ void rope_kernel(_Float16* __restrict__ qg, _Float16* __restrict__ kg) {
  size_t idx = (size_t)blockIdx.x * blockDim.x + threadIdx.x;
  if (idx >= (size_t)B_ * K_ * H_ * CAP_ * 32) return;
  int i = (int)(idx & 31);
  size_t rest = idx >> 5;          // head*CAP + c
  int c = (int)(rest % CAP_);
  size_t head = rest / CAP_;
  float inv = __powf(10000.0f, -(float)i / 32.0f);
  float ang = (float)c * inv, cs, sn;
  __sincosf(ang, &cs, &sn);

  size_t qb = rest * HD_;
  float q1 = (float)qg[qb + i], q2 = (float)qg[qb + i + 32];
  qg[qb + i]      = (_Float16)(q1 * cs - q2 * sn);
  qg[qb + i + 32] = (_Float16)(q2 * cs + q1 * sn);

  size_t kb = head * (HD_ * CAP_) + c;
  float k1 = (float)kg[kb + (size_t)i * CAP_], k2 = (float)kg[kb + (size_t)(i + 32) * CAP_];
  kg[kb + (size_t)i * CAP_]        = (_Float16)(k1 * cs - k2 * sn);
  kg[kb + (size_t)(i + 32) * CAP_] = (_Float16)(k2 * cs + k1 * sn);
}

// ---------------------------------------------------------------------------
// Flash attention per (b,k,h): wave owns 16 query rows, 32-key chunks,
// 8 WMMAs/chunk. K is pre-transposed -> K/V B-fragments are per-element loads,
// each 32B contiguous across a 16-lane half. grid = (CAP/64, B*K*H), block = 128 (4 waves).
// ---------------------------------------------------------------------------
__global__ void __launch_bounds__(128) attn_kernel(
    const _Float16* __restrict__ qg, const _Float16* __restrict__ kg,
    const _Float16* __restrict__ vg, const int* __restrict__ topk,
    const int* __restrict__ Lbuf, float* __restrict__ out_sum) {
  int wave = threadIdx.x >> 5, lane = threadIdx.x & 31;
  int m = lane & 15, half = lane >> 4;
  int by = blockIdx.y;
  int h = by % H_, bk = by / H_, k = bk % K_, b = bk / K_;
  int L = Lbuf[b * K_ + k];
  if (L == 0) return;
  int q0 = blockIdx.x * 64 + wave * 16;

  size_t hbase = (((size_t)(b * K_ + k)) * H_ + h) * CAP_ * HD_;
  const _Float16* Q   = qg + hbase;
  const _Float16* KmT = kg + hbase;   // (hd, c) layout
  const _Float16* Vm  = vg + hbase;   // (c, hd) layout

  __shared__ _Float16 pls[4][16][32];

  v16h qa[2];
#pragma unroll
  for (int t = 0; t < 2; ++t) {
    const _Float16* qp = Q + (size_t)(q0 + m) * HD_ + t * 32 + (half ? 8 : 0);
#pragma unroll
    for (int i = 0; i < 8; ++i) { qa[t][i] = qp[i]; qa[t][i + 8] = qp[i + 16]; }
  }

  float mrow[8], lrow[8];
#pragma unroll
  for (int r = 0; r < 8; ++r) { mrow[r] = -INFINITY; lrow[r] = 0.f; }
  v8f O[4] = {};

  int climit = q0 + 15;
  for (int c0 = 0; c0 <= climit; c0 += 32) {
    v8f s0 = {}, s1 = {};
#pragma unroll
    for (int t = 0; t < 2; ++t) {
      int kd = t * 32 + half * 8;       // hd base for this lane half
      v16h b0, b1;                      // key c0+m / c0+16+m
#pragma unroll
      for (int i = 0; i < 16; ++i) {
        const _Float16* kp = KmT + (size_t)(kd + (i & 7) + (i >> 3) * 16) * CAP_ + c0 + m;
        b0[i] = kp[0];
        b1[i] = kp[16];
      }
      s0 = WMMA_F16(qa[t], b0, s0);
      s1 = WMMA_F16(qa[t], b1, s1);
    }

    float mx[8], fac[8], p0a[8], p1a[8], ps[8];
#pragma unroll
    for (int r = 0; r < 8; ++r) {
      int q = q0 + r + half * 8;
      int cA = c0 + m, cB = c0 + 16 + m;
      float sa = s0[r] * SCALE_, sb = s1[r] * SCALE_;
      if (cA > q || cA >= L) sa = NEG_;
      if (cB > q || cB >= L) sb = NEG_;
      s0[r] = sa; s1[r] = sb;
      mx[r] = fmaxf(sa, sb);
    }
#pragma unroll
    for (int off = 8; off; off >>= 1)
#pragma unroll
      for (int r = 0; r < 8; ++r) mx[r] = fmaxf(mx[r], __shfl_xor(mx[r], off, 32));
#pragma unroll
    for (int r = 0; r < 8; ++r) {
      float mn = fmaxf(mrow[r], mx[r]);
      fac[r] = __expf(mrow[r] - mn);
      p0a[r] = __expf(s0[r] - mn);
      p1a[r] = __expf(s1[r] - mn);
      mrow[r] = mn;
      ps[r] = p0a[r] + p1a[r];
    }
#pragma unroll
    for (int off = 8; off; off >>= 1)
#pragma unroll
      for (int r = 0; r < 8; ++r) ps[r] += __shfl_xor(ps[r], off, 32);
#pragma unroll
    for (int r = 0; r < 8; ++r) lrow[r] = lrow[r] * fac[r] + ps[r];
#pragma unroll
    for (int n0 = 0; n0 < 4; ++n0)
#pragma unroll
      for (int r = 0; r < 8; ++r) O[n0][r] *= fac[r];

    // re-fragment P: C-layout -> A-layout through per-wave LDS (no barrier:
    // LDS ops are in-order within a wave)
#pragma unroll
    for (int r = 0; r < 8; ++r) {
      pls[wave][r + half * 8][m]      = (_Float16)p0a[r];
      pls[wave][r + half * 8][m + 16] = (_Float16)p1a[r];
    }
    v16h pa;
    {
      int koff = half ? 8 : 0;
#pragma unroll
      for (int i = 0; i < 8; ++i) {
        pa[i]     = pls[wave][m][koff + i];
        pa[i + 8] = pls[wave][m][16 + koff + i];
      }
    }

    const _Float16* vp = Vm + (size_t)(c0 + half * 8) * HD_ + m;
#pragma unroll
    for (int n0 = 0; n0 < 4; ++n0) {
      v16h vb;
#pragma unroll
      for (int i = 0; i < 16; ++i) vb[i] = vp[(size_t)((i & 7) + (i >> 3) * 16) * HD_ + n0 * 16];
      O[n0] = WMMA_F16(pa, vb, O[n0]);
    }
  }

  const int* ti = topk + (b * K_ + k) * CAP_;
#pragma unroll
  for (int r = 0; r < 8; ++r) {
    int q = q0 + r + half * 8;
    int token = ti[q];
    float inv = 1.0f / lrow[r];
    float* dst = out_sum + ((size_t)(b * N_ + token)) * (H_ * HD_) + h * HD_;
#pragma unroll
    for (int n0 = 0; n0 < 4; ++n0)
      atomicAdd(dst + n0 * 16 + m, O[n0][r] * inv);
  }
}

// ---------------------------------------------------------------------------
// Counts, normalize (f32 -> f16 for final GEMM), aux-loss finalize.
// ---------------------------------------------------------------------------
__global__ void counts_kernel(const int* __restrict__ topk, float* __restrict__ counts) {
  int i = blockIdx.x * blockDim.x + threadIdx.x;
  if (i >= B_ * K_ * CAP_) return;
  int b = i / (K_ * CAP_);
  atomicAdd(&counts[b * N_ + topk[i]], 1.0f);
}

__global__ void normalize_kernel(const float* __restrict__ out_sum,
                                 const float* __restrict__ counts,
                                 _Float16* __restrict__ mh) {
  size_t i = (size_t)blockIdx.x * blockDim.x + threadIdx.x;
  if (i >= (size_t)B_ * N_ * H_ * HD_) return;
  size_t bn = i / (H_ * HD_);
  float c = counts[bn];
  mh[i] = (_Float16)(out_sum[i] / fmaxf(c, 1.0f));
}

__global__ void aux_kernel(const float* __restrict__ cnt_bk,
                           const float* __restrict__ psum_bk, float* __restrict__ dst) {
  float acc = 0.f;
  for (int b = 0; b < B_; ++b) {
    float sb = 0.f;
    for (int e = 0; e < K_; ++e)
      sb += (cnt_bk[b * K_ + e] / (float)N_) * (psum_bk[b * K_ + e] / (float)N_);
    acc += sb;
  }
  float aux = (float)K_ * acc / (float)B_;
  dst[0] = fminf(fmaxf(aux, 0.0f), 10.0f);
}

// ---------------------------------------------------------------------------
// Final GEMM: mh (f16, row-major) @ Wo (f32, LDS-staged) -> d_out (f32).
// ---------------------------------------------------------------------------
__global__ void __launch_bounds__(256) out_gemm(
    const _Float16* __restrict__ A, const float* __restrict__ W,
    float* __restrict__ Cout) {
  __shared__ float wtile[32][65];
  int wave = threadIdx.x >> 5, lane = threadIdx.x & 31;
  int m = lane & 15, half = lane >> 4;
  int row0 = blockIdx.y * 128 + wave * 16;
  int col0 = blockIdx.x * 64;
  const _Float16* arow = A + (size_t)(row0 + m) * (H_ * HD_);

  v8f acc[4] = {};
  for (int kk = 0; kk < H_ * HD_; kk += 32) {
    {
      int r  = threadIdx.x >> 3;
      int c8 = (threadIdx.x & 7) * 8;
      const float* wp = W + (size_t)(kk + r) * D_ + col0 + c8;
      __builtin_prefetch(wp + 32 * D_, 0, 0);
#pragma unroll
      for (int i = 0; i < 8; ++i) wtile[r][c8 + i] = wp[i];
    }
    __syncthreads();

    v16h a;
    const _Float16* ap = arow + kk + (half ? 8 : 0);
#pragma unroll
    for (int i = 0; i < 8; ++i) { a[i] = ap[i]; a[i + 8] = ap[i + 16]; }

#pragma unroll
    for (int n = 0; n < 4; ++n) {
      v16h bf;
#pragma unroll
      for (int i = 0; i < 16; ++i) bf[i] = (_Float16)wtile[half * 8 + (i & 7) + (i >> 3) * 16][n * 16 + m];
      acc[n] = WMMA_F16(a, bf, acc[n]);
    }
    __syncthreads();
  }
#pragma unroll
  for (int n = 0; n < 4; ++n)
#pragma unroll
    for (int r = 0; r < 8; ++r)
      Cout[(size_t)(row0 + r + half * 8) * D_ + col0 + n * 16 + m] = acc[n][r];
}

// ---------------------------------------------------------------------------
extern "C" void kernel_launch(void* const* d_in, const int* in_sizes, int n_in,
                              void* d_out, int out_size, void* d_ws, size_t ws_size,
                              hipStream_t stream) {
  (void)in_sizes; (void)n_in; (void)out_size; (void)ws_size;
  const float* x  = (const float*)d_in[0];
  const float* Wq = (const float*)d_in[1];
  const float* Wk = (const float*)d_in[2];
  const float* Wv = (const float*)d_in[3];
  const float* Wo = (const float*)d_in[4];
  const float* Wr = (const float*)d_in[5];
  float* out = (float*)d_out;

  char* ws = (char*)d_ws;
  size_t off = 0;
  auto take = [&](size_t bytes) -> char* {
    char* p = ws + off;
    off += (bytes + 255) & ~(size_t)255;
    return p;
  };
  float*     probs   = (float*)take((size_t)B_ * N_ * K_ * 4);
  int*       choice  = (int*)  take((size_t)B_ * N_ * 4);
  float*     cnt_bk  = (float*)take(B_ * K_ * 4);
  float*     psum_bk = (float*)take(B_ * K_ * 4);
  int*       Lbuf    = (int*)  take(B_ * K_ * 4);
  int*       topk    = (int*)  take((size_t)B_ * K_ * CAP_ * 4);
  _Float16*  qg      = (_Float16*)take((size_t)B_ * K_ * H_ * CAP_ * HD_ * 2);
  _Float16*  kg      = (_Float16*)take((size_t)B_ * K_ * H_ * CAP_ * HD_ * 2);
  _Float16*  vg      = (_Float16*)take((size_t)B_ * K_ * H_ * CAP_ * HD_ * 2);
  float*     out_sum = (float*)take((size_t)B_ * N_ * H_ * HD_ * 4);
  float*     counts  = (float*)take((size_t)B_ * N_ * 4);
  _Float16*  mh      = qg;   // qg dead after attention; reuse its 16 MB

  hipMemsetAsync(out_sum, 0, (size_t)B_ * N_ * H_ * HD_ * 4, stream);
  hipMemsetAsync(counts, 0, (size_t)B_ * N_ * 4, stream);
  hipMemsetAsync(cnt_bk, 0, B_ * K_ * 4, stream);
  hipMemsetAsync(psum_bk, 0, B_ * K_ * 4, stream);
  hipMemsetAsync(Lbuf, 0, B_ * K_ * 4, stream);

  router_kernel<<<(B_ * N_) / 8, 256, 0, stream>>>(x, Wr, probs, choice, cnt_bk, psum_bk);
  topk_sort_kernel<<<B_ * K_, 1024, 0, stream>>>(probs, choice, topk, Lbuf);

  dim3 gg(16, (B_ * K_ * CAP_) / 128);
  qkv_gather_gemm<false><<<gg, 256, 0, stream>>>(x, Wq, topk, qg);
  qkv_gather_gemm<true ><<<gg, 256, 0, stream>>>(x, Wk, topk, kg);  // K transposed
  qkv_gather_gemm<false><<<gg, 256, 0, stream>>>(x, Wv, topk, vg);

  size_t rope_threads = (size_t)B_ * K_ * H_ * CAP_ * 32;
  rope_kernel<<<(unsigned)((rope_threads + 255) / 256), 256, 0, stream>>>(qg, kg);

  attn_kernel<<<dim3(CAP_ / 64, B_ * K_ * H_), 128, 0, stream>>>(qg, kg, vg, topk, Lbuf, out_sum);

  counts_kernel<<<(B_ * K_ * CAP_ + 255) / 256, 256, 0, stream>>>(topk, counts);

  size_t nelem = (size_t)B_ * N_ * H_ * HD_;
  normalize_kernel<<<(unsigned)((nelem + 255) / 256), 256, 0, stream>>>(out_sum, counts, mh);

  out_gemm<<<dim3(D_ / 64, (B_ * N_) / 128), 256, 0, stream>>>(mh, Wo, out);

  aux_kernel<<<1, 1, 0, stream>>>(cnt_bk, psum_bk, out + (size_t)B_ * N_ * D_);
}
